// RPPatchEmbed_no_share_61443802136907
// MI455X (gfx1250) — matrix operand order repaired and checked by reading, not compiled, not run on hardware
//
#include <hip/hip_runtime.h>

// ---------------------------------------------------------------------------
// RPPatchEmbed pipeline for gfx1250 (MI455X), wave32, WMMA bf16 16x16x32.
// Zero-halo padded activations -> branchless im2col gather in the GEMM loop.
// ---------------------------------------------------------------------------

typedef __attribute__((ext_vector_type(16))) __bf16 v16bf;
typedef __attribute__((ext_vector_type(8)))  float  v8f;

#define EPSF 1e-5f

// ------------------------- small utility kernels ---------------------------

// memset-style zero fill (u32 granularity)
__global__ __launch_bounds__(256) void zero_u32(unsigned* __restrict__ p, long n) {
  long i = (long)blockIdx.x * 256 + threadIdx.x;
  if (i < n) p[i] = 0u;
}

// x: (32,3,224,224) fp32 NCHW -> interior of xbfp: (32,256,256,3) bf16 NHWC,
// halo = 16 (covers conv1 pad 1 and proj pads 0/8/16).
__global__ __launch_bounds__(256) void x_to_bf16(const float* __restrict__ x,
                                                 __bf16* __restrict__ xbfp) {
  int idx = blockIdx.x * 256 + threadIdx.x;
  const int total = 32 * 224 * 224 * 3;
  if (idx >= total) return;
  int c = idx % 3;
  int r = idx / 3;
  int w = r % 224; r /= 224;
  int h = r % 224;
  int b = r / 224;
  xbfp[(((size_t)b * 256 + h + 16) * 256 + (w + 16)) * 3 + c] =
      (__bf16)x[(((size_t)b * 3 + c) * 224 + h) * 224 + w];
}

// weights (Cout,Cin,ks,ks) fp32 -> bf16 [Kpad][Npad], k = (kh*ks+kw)*Cin + c
__global__ __launch_bounds__(256) void wt_to_bf16(const float* __restrict__ src,
                                                  __bf16* __restrict__ dst,
                                                  int Cout, int Cin, int ks,
                                                  int Kact, int Kpad, int Npad) {
  size_t total = (size_t)Kpad * Npad;
  size_t i = (size_t)blockIdx.x * 256 + threadIdx.x;
  if (i >= total) return;
  int e = (int)(i % Npad);
  int k = (int)(i / Npad);
  float v = 0.f;
  if (k < Kact && e < Cout) {
    int c = k % Cin;
    int khkw = k / Cin;
    int kw = khkw % ks;
    int kh = khkw / ks;
    v = src[(((size_t)e * Cin + c) * ks + kh) * ks + kw];
  }
  dst[i] = (__bf16)v;
}

// im2col direct-offset table: tab[k] = (kh*IWp + kw)*Cin + c  (element offset
// relative to the gather row origin in the PADDED tensor); -1 for pad K.
__global__ __launch_bounds__(256) void build_ktab(int* __restrict__ tab,
                                                  int Cin, int ks, int IWp,
                                                  int Kact, int Kpad) {
  int k = blockIdx.x * 256 + threadIdx.x;
  if (k >= Kpad) return;
  if (k >= Kact) { tab[k] = -1; return; }
  int c = k % Cin;
  int khkw = k / Cin;
  int kw = khkw % ks;
  int kh = khkw / ks;
  tab[k] = (kh * IWp + kw) * Cin + c;
}

// Fold conv bias + BN into per-channel scale/shift.
// Layout (floats): conv1 s@0,h@64 | conv2 s@128,h@256 | conv3 s@384,h@640 |
//                  conv4 s@896,h@960 | proj_i s@1024+i*1536, h@+768
__global__ __launch_bounds__(256) void fuse_params(
    const float* rb1, const float* rb2, const float* rb3, const float* rb4,
    const float* g1, const float* b1, const float* m1, const float* v1,
    const float* g2, const float* b2, const float* m2, const float* v2,
    const float* g3, const float* b3, const float* m3, const float* v3,
    const float* pb0, const float* pb1, const float* pb2,
    float* __restrict__ fused) {
  int tid = threadIdx.x;
  for (int c = tid; c < 64; c += 256) {
    float inv = g1[c] / sqrtf(v1[c] + EPSF);
    fused[0 + c]  = inv;
    fused[64 + c] = rb1[c] * inv + b1[c] - m1[c] * inv;
  }
  for (int c = tid; c < 128; c += 256) {
    float inv = g2[c] / sqrtf(v2[c] + EPSF);
    fused[128 + c] = inv;
    fused[256 + c] = rb2[c] * inv + b2[c] - m2[c] * inv;
  }
  for (int c = tid; c < 256; c += 256) {
    float inv = g3[c] / sqrtf(v3[c] + EPSF);
    fused[384 + c] = inv;
    fused[640 + c] = rb3[c] * inv + b3[c] - m3[c] * inv;
  }
  for (int c = tid; c < 64; c += 256) {
    fused[896 + c] = 1.f;
    fused[960 + c] = (c < 3) ? rb4[c] : 0.f;
  }
  const float* pbs[3] = {pb0, pb1, pb2};
  for (int i = 0; i < 3; ++i)
    for (int c = tid; c < 768; c += 256) {
      fused[1024 + i * 1536 + c]       = 1.f;
      fused[1024 + i * 1536 + 768 + c] = pbs[i][c];
    }
}

// ------------------------- implicit-GEMM WMMA conv --------------------------
//
// out[m, e] = act( (sum_k im2col(in)[m,k] * wt[k,e]) * scale[e] + shift[e] )
// in: PADDED NHWC bf16 (zero halo, no bounds checks).  wt: [Kpad][Npad] bf16.
// Block = 128 threads (4 waves). WG tile: 16 rows x (NT*64) cols.
// Each wave computes NT 16x16 D tiles with v_wmma_f32_16x16x32_bf16.
// OUTMODE: 0 = padded NHWC bf16 activation interior (optional relu)
//          1 = fp32 embed rows, out[m*768 + e]
//          2 = fp32 score, out[b*588 + e*196 + pos]

struct ConvGeom {
  int Cin, IHp, IWp;        // padded input dims
  int OH, OW;               // logical output dims
  int stride, pad, inHalo;  // gather origin = o*stride - pad + inHalo (>= 0)
  int Kpad, Nact, Npad;
  int OHp, OWp, oHalo;      // padded output dims (OUTMODE 0)
  int relu;
};

template <int OUTMODE, int NT>
__global__ __launch_bounds__(128) void conv_gemm(
    const __bf16* __restrict__ in, const __bf16* __restrict__ wt,
    const int* __restrict__ ktab,
    const float* __restrict__ scale, const float* __restrict__ shift,
    float* __restrict__ outF, __bf16* __restrict__ outB, ConvGeom g) {
  constexpr int WGN = NT * 64;           // columns covered by this workgroup
  __shared__ __bf16 As[16][34];          // 16 rows x 32 k (odd stride: no conflicts)
  __shared__ __bf16 Bs[WGN][34];         // transposed [n][k]

  const int tid  = threadIdx.x;
  const int wave = tid >> 5;
  const int lane = tid & 31;
  const int lgrp = lane >> 4;            // 0 | 1 (lane group)
  const int lrow = lane & 15;            // M index (A) / N index (B,D)

  const int mTile = blockIdx.x;
  const int nBase = blockIdx.y * WGN;

  // per-thread A-staging coordinates: row = tid/8, 4 consecutive k
  const int aRow = tid >> 3;
  const int aK0  = (tid & 7) * 4;
  const int ohw  = g.OH * g.OW;
  const int mA   = mTile * 16 + aRow;
  const int bIdxA = mA / ohw;
  const int posA  = mA - bIdxA * ohw;
  const int ohA   = posA / g.OW;
  const int owA   = posA - ohA * g.OW;
  const int orgH  = ohA * g.stride - g.pad + g.inHalo;   // always >= 0
  const int orgW  = owA * g.stride - g.pad + g.inHalo;   // always >= 0
  const __bf16* rp =
      in + ((size_t)(bIdxA * g.IHp + orgH) * g.IWp + orgW) * g.Cin;

  v8f acc[NT];
#pragma unroll
  for (int t = 0; t < NT; ++t) acc[t] = (v8f)0.0f;

  for (int k0 = 0; k0 < g.Kpad; k0 += 32) {
    // ---- stage A: branchless gather via direct-offset table ----
    {
      int4 d4 = *(const int4*)(ktab + k0 + aK0);
      int doffs[4] = {d4.x, d4.y, d4.z, d4.w};
#pragma unroll
      for (int jj = 0; jj < 4; ++jj) {
        int doff = doffs[jj];
        __bf16 val = rp[doff < 0 ? 0 : doff];        // halo is zero; clamp pad-K
        As[aRow][aK0 + jj] = (doff < 0) ? (__bf16)0.0f : val;
      }
    }
    // ---- stage B (coalesced rows of wt, transpose into LDS) ----
    {
      constexpr int NVEC = WGN >> 2;             // 4-wide groups per k-row
      constexpr int ITER = (32 * NVEC) / 128;    // per-thread iterations
#pragma unroll
      for (int it = 0; it < ITER; ++it) {
        int idx = tid + it * 128;
        int kr  = idx / NVEC;
        int nl4 = (idx - kr * NVEC) << 2;
        const __bf16* wrow = wt + (size_t)(k0 + kr) * g.Npad + nBase + nl4;
        __builtin_prefetch(wrow + (size_t)32 * g.Npad, 0, 1);  // next K-step
        __bf16 w0 = wrow[0], w1 = wrow[1], w2 = wrow[2], w3 = wrow[3];
        Bs[nl4 + 0][kr] = w0;
        Bs[nl4 + 1][kr] = w1;
        Bs[nl4 + 2][kr] = w2;
        Bs[nl4 + 3][kr] = w3;
      }
    }
    __syncthreads();

    // ---- A fragment (16-bit A 16x32 layout, ISA 7.12.2) ----
    v16bf af;
#pragma unroll
    for (int h = 0; h < 16; ++h) {
      int v = h >> 1, p = h & 1;
      int kk = ((v & 4) ? 16 : 0) + lgrp * 8 + (v & 3) * 2 + p;
      af[h] = As[lrow][kk];
    }
    // ---- B fragments + WMMA accumulate ----
#pragma unroll
    for (int t = 0; t < NT; ++t) {
      int nLocal = (wave * NT + t) * 16 + lrow;
      v16bf bfrag;
#pragma unroll
      for (int h = 0; h < 16; ++h) {
        int v = h >> 1, p = h & 1;
        int kk = ((v & 4) ? 16 : 0) + lgrp * 8 + (v & 3) * 2 + p;
        bfrag[h] = Bs[nLocal][kk];
      }
      acc[t] = __builtin_amdgcn_wmma_f32_16x16x32_bf16(
          false, af, false, bfrag, (short)0, acc[t], false, false);
    }
    __syncthreads();
  }

  // ---- epilogue: scale/shift (+relu), store per OUTMODE ----
  const int mBase = mTile * 16 + lgrp * 8;
#pragma unroll
  for (int t = 0; t < NT; ++t) {
    int e = nBase + (wave * NT + t) * 16 + lrow;
    if (e >= g.Nact) continue;
    float sc = scale[e];
    float sh = shift[e];
#pragma unroll
    for (int v = 0; v < 8; ++v) {
      int m = mBase + v;
      float val = acc[t][v] * sc + sh;
      if (g.relu) val = fmaxf(val, 0.f);
      if (OUTMODE == 0) {
        int bI = m / ohw;
        int pos = m - bI * ohw;
        int oh = pos / g.OW;
        int ow = pos - oh * g.OW;
        outB[(((size_t)bI * g.OHp + oh + g.oHalo) * g.OWp + ow + g.oHalo) *
                 g.Nact + e] = (__bf16)val;
      } else if (OUTMODE == 1) {
        outF[(size_t)m * 768 + e] = val;
      } else {
        int bI = m / 196;
        int pos = m - bI * 196;
        outF[(size_t)bI * 588 + e * 196 + pos] = val;
      }
    }
  }
}

// ------------------------- top-K / softmax ----------------------------------
// One block per batch. Bitonic sort of 1024 (score,idx) padded with -inf.
// Ascending => real 588 entries end up in slots 436..1023; top 196 = 828..1023
// in ascending score order, matching order[:, J:].
__global__ __launch_bounds__(1024) void topk_softmax(const float* __restrict__ score,
                                                     int* __restrict__ topIdx,
                                                     float* __restrict__ topW) {
  __shared__ float sv[1024];
  __shared__ int   si[1024];
  __shared__ float red[1024];
  const int tid = threadIdx.x;
  const int b   = blockIdx.x;

  sv[tid] = (tid < 588) ? score[b * 588 + tid] : -__builtin_inff();
  si[tid] = tid;
  __syncthreads();

  for (int k = 2; k <= 1024; k <<= 1) {
    for (int j = k >> 1; j > 0; j >>= 1) {
      int ixj = tid ^ j;
      if (ixj > tid) {
        bool asc = ((tid & k) == 0);
        float a = sv[tid], c = sv[ixj];
        if ((a > c) == asc) {
          sv[tid] = c; sv[ixj] = a;
          int t2 = si[tid]; si[tid] = si[ixj]; si[ixj] = t2;
        }
      }
      __syncthreads();
    }
  }

  float maxv = sv[1023];
  red[tid] = (tid >= 436) ? __expf(sv[tid] - maxv) : 0.f;
  __syncthreads();
  for (int s = 512; s > 0; s >>= 1) {
    if (tid < s) red[tid] += red[tid + s];
    __syncthreads();
  }
  float sum = red[0];
  if (tid < 196) {
    int p = 828 + tid;
    float smn = __expf(sv[p] - maxv) / sum;
    topIdx[b * 196 + tid] = si[p];
    topW[b * 196 + tid]   = (1.0f - smn) + smn;   // faithful STE forward
  }
}

// merged[b, j, :] = embed[a, b*196+pos, :] * w   (n = a*196+pos)
__global__ __launch_bounds__(256) void gather_merge(const float* __restrict__ embed,
                                                    const int* __restrict__ topIdx,
                                                    const float* __restrict__ topW,
                                                    float* __restrict__ out) {
  int blk = blockIdx.x;                  // b*196 + j
  int b = blk / 196;
  int n = topIdx[blk];
  float w = topW[blk];
  int a = n / 196;
  int pos = n - a * 196;
  const float* src = embed + ((size_t)a * 6272 + b * 196 + pos) * 768;
  float* dst = out + (size_t)blk * 768;
  for (int e = threadIdx.x; e < 768; e += 256) dst[e] = src[e] * w;
}

// ------------------------------ launcher ------------------------------------

extern "C" void kernel_launch(void* const* d_in, const int* in_sizes, int n_in,
                              void* d_out, int out_size, void* d_ws, size_t ws_size,
                              hipStream_t stream) {
  const float* x   = (const float*)d_in[0];
  const float* pw[3] = {(const float*)d_in[1], (const float*)d_in[3], (const float*)d_in[5]};
  const float* pb[3] = {(const float*)d_in[2], (const float*)d_in[4], (const float*)d_in[6]};
  const float* rw[4] = {(const float*)d_in[7], (const float*)d_in[9],
                        (const float*)d_in[11], (const float*)d_in[13]};
  const float* rb[4] = {(const float*)d_in[8], (const float*)d_in[10],
                        (const float*)d_in[12], (const float*)d_in[14]};
  const float* bn[3][4];
  for (int i = 0; i < 3; ++i)
    for (int j = 0; j < 4; ++j) bn[i][j] = (const float*)d_in[15 + i * 4 + j];

  // ---- workspace layout (padded activation buffers) ----
  char* ws = (char*)d_ws;
  size_t off = 0;
  auto alloc = [&](size_t bytes) {
    size_t o = off;
    off = (off + bytes + 255) & ~(size_t)255;
    return o;
  };
  const size_t nXBFP = (size_t)32 * 256 * 256 * 3;     // halo 16
  const size_t nR1P  = (size_t)32 * 114 * 114 * 64;    // halo 1
  const size_t nR2P  = (size_t)32 * 58 * 58 * 128;     // halo 1
  const size_t nR3P  = (size_t)32 * 30 * 30 * 256;     // halo 1
  size_t o_xbfp  = alloc(nXBFP * 2);
  size_t o_wp0   = alloc((size_t)768 * 768 * 2);
  size_t o_wp1   = alloc((size_t)3072 * 768 * 2);
  size_t o_wp2   = alloc((size_t)6912 * 768 * 2);
  size_t o_wr1   = alloc((size_t)32 * 64 * 2);
  size_t o_wr2   = alloc((size_t)576 * 128 * 2);
  size_t o_wr3   = alloc((size_t)1152 * 256 * 2);
  size_t o_wr4   = alloc((size_t)2304 * 64 * 2);
  size_t o_fused = alloc((size_t)5632 * 4);
  size_t o_r1p   = alloc(nR1P * 2);
  size_t o_r2p   = alloc(nR2P * 2);
  size_t o_r3p   = alloc(nR3P * 2);
  size_t o_score = alloc((size_t)32 * 588 * 4);
  size_t o_embed = alloc((size_t)3 * 6272 * 768 * 4);
  size_t o_tidx  = alloc((size_t)32 * 196 * 4);
  size_t o_tw    = alloc((size_t)32 * 196 * 4);
  size_t o_kt_r1 = alloc((size_t)32 * 4);
  size_t o_kt_r2 = alloc((size_t)576 * 4);
  size_t o_kt_r3 = alloc((size_t)1152 * 4);
  size_t o_kt_r4 = alloc((size_t)2304 * 4);
  size_t o_kt_p0 = alloc((size_t)768 * 4);
  size_t o_kt_p1 = alloc((size_t)3072 * 4);
  size_t o_kt_p2 = alloc((size_t)6912 * 4);
  (void)ws_size; (void)n_in; (void)in_sizes; (void)out_size;

  __bf16* xbfp  = (__bf16*)(ws + o_xbfp);
  __bf16* wpB[3] = {(__bf16*)(ws + o_wp0), (__bf16*)(ws + o_wp1), (__bf16*)(ws + o_wp2)};
  __bf16* wrB[4] = {(__bf16*)(ws + o_wr1), (__bf16*)(ws + o_wr2),
                    (__bf16*)(ws + o_wr3), (__bf16*)(ws + o_wr4)};
  float*  fused = (float*)(ws + o_fused);
  __bf16* r1p = (__bf16*)(ws + o_r1p);
  __bf16* r2p = (__bf16*)(ws + o_r2p);
  __bf16* r3p = (__bf16*)(ws + o_r3p);
  float*  score = (float*)(ws + o_score);
  float*  embed = (float*)(ws + o_embed);
  int*    topIdx = (int*)(ws + o_tidx);
  float*  topW   = (float*)(ws + o_tw);
  int* ktR[4] = {(int*)(ws + o_kt_r1), (int*)(ws + o_kt_r2),
                 (int*)(ws + o_kt_r3), (int*)(ws + o_kt_r4)};
  int* ktP[3] = {(int*)(ws + o_kt_p0), (int*)(ws + o_kt_p1), (int*)(ws + o_kt_p2)};
  float*  out = (float*)d_out;

  // ---- zero padded activation buffers (halos must be 0 every call) ----
  {
    long n;
    n = (long)(nXBFP / 2);
    zero_u32<<<(unsigned)((n + 255) / 256), 256, 0, stream>>>((unsigned*)xbfp, n);
    n = (long)(nR1P / 2);
    zero_u32<<<(unsigned)((n + 255) / 256), 256, 0, stream>>>((unsigned*)r1p, n);
    n = (long)(nR2P / 2);
    zero_u32<<<(unsigned)((n + 255) / 256), 256, 0, stream>>>((unsigned*)r2p, n);
    n = (long)(nR3P / 2);
    zero_u32<<<(unsigned)((n + 255) / 256), 256, 0, stream>>>((unsigned*)r3p, n);
  }

  // ---- conversions ----
  {
    int total = 32 * 224 * 224 * 3;
    x_to_bf16<<<(total + 255) / 256, 256, 0, stream>>>(x, xbfp);
  }
  // proj weights: Cout=768, Cin=3, ks=16*(a+1), K = 3*ks^2
  const int pks[3] = {16, 32, 48};
  const int pK[3]  = {768, 3072, 6912};
  for (int a = 0; a < 3; ++a) {
    size_t tot = (size_t)pK[a] * 768;
    wt_to_bf16<<<(int)((tot + 255) / 256), 256, 0, stream>>>(
        pw[a], wpB[a], 768, 3, pks[a], pK[a], pK[a], 768);
    build_ktab<<<(pK[a] + 255) / 256, 256, 0, stream>>>(
        ktP[a], 3, pks[a], 256, pK[a], pK[a]);
  }
  // rpn weights
  const int rCout[4] = {64, 128, 256, 3};
  const int rCin[4]  = {3, 64, 128, 256};
  const int rKact[4] = {27, 576, 1152, 2304};
  const int rKpad[4] = {32, 576, 1152, 2304};
  const int rNpad[4] = {64, 128, 256, 64};
  const int rIWp[4]  = {256, 114, 58, 30};   // padded input width per rpn conv
  for (int i = 0; i < 4; ++i) {
    size_t tot = (size_t)rKpad[i] * rNpad[i];
    wt_to_bf16<<<(int)((tot + 255) / 256), 256, 0, stream>>>(
        rw[i], wrB[i], rCout[i], rCin[i], 3, rKact[i], rKpad[i], rNpad[i]);
    build_ktab<<<(rKpad[i] + 255) / 256, 256, 0, stream>>>(
        ktR[i], rCin[i], 3, rIWp[i], rKact[i], rKpad[i]);
  }
  fuse_params<<<1, 256, 0, stream>>>(
      rb[0], rb[1], rb[2], rb[3],
      bn[0][0], bn[0][1], bn[0][2], bn[0][3],
      bn[1][0], bn[1][1], bn[1][2], bn[1][3],
      bn[2][0], bn[2][1], bn[2][2], bn[2][3],
      pb[0], pb[1], pb[2], fused);

  // ---- RPN via implicit-GEMM WMMA ----
  // {Cin, IHp, IWp, OH, OW, stride, pad, inHalo, Kpad, Nact, Npad, OHp, OWp, oHalo, relu}
  ConvGeom g1 = {3, 256, 256, 112, 112, 2, 1, 16, 32, 64, 64, 114, 114, 1, 1};
  conv_gemm<0, 1><<<dim3(401408 / 16, 1), 128, 0, stream>>>(
      xbfp, wrB[0], ktR[0], fused + 0, fused + 64, nullptr, r1p, g1);

  ConvGeom g2 = {64, 114, 114, 56, 56, 2, 1, 1, 576, 128, 128, 58, 58, 1, 1};
  conv_gemm<0, 2><<<dim3(100352 / 16, 1), 128, 0, stream>>>(
      r1p, wrB[1], ktR[1], fused + 128, fused + 256, nullptr, r2p, g2);

  ConvGeom g3 = {128, 58, 58, 28, 28, 2, 1, 1, 1152, 256, 256, 30, 30, 1, 1};
  conv_gemm<0, 4><<<dim3(25088 / 16, 1), 128, 0, stream>>>(
      r2p, wrB[2], ktR[2], fused + 384, fused + 640, nullptr, r3p, g3);

  ConvGeom g4 = {256, 30, 30, 14, 14, 2, 1, 1, 2304, 3, 64, 0, 0, 0, 0};
  conv_gemm<2, 1><<<dim3(6272 / 16, 1), 128, 0, stream>>>(
      r3p, wrB[3], ktR[3], fused + 896, fused + 960, score, nullptr, g4);

  // ---- patch-embed GEMMs via WMMA ----
  const int ppad[3] = {0, 8, 16};
  for (int a = 0; a < 3; ++a) {
    ConvGeom gp = {3, 256, 256, 14, 14, 16, ppad[a], 16,
                   pK[a], 768, 768, 0, 0, 0, 0};
    conv_gemm<1, 4><<<dim3(6272 / 16, 3), 128, 0, stream>>>(
        xbfp, wpB[a], ktP[a], fused + 1024 + a * 1536,
        fused + 1024 + a * 1536 + 768, embed + (size_t)a * 6272 * 768,
        nullptr, gp);
  }

  // ---- top-K + softmax STE weights ----
  topk_softmax<<<32, 1024, 0, stream>>>(score, topIdx, topW);

  // ---- gather & scale into output (32,196,768) fp32 ----
  gather_merge<<<32 * 196, 256, 0, stream>>>(embed, topIdx, topW, out);
}
